// LengthRegulator_19834158973174
// MI455X (gfx1250) — matrix operand order repaired and checked by reading, not compiled, not run on hardware
//
#include <hip/hip_runtime.h>

typedef _Float16 v16h __attribute__((ext_vector_type(16)));
typedef float    v8f  __attribute__((ext_vector_type(8)));
typedef unsigned int v4u __attribute__((ext_vector_type(4)));

#define NB   16      // batch
#define LL   512     // sequence length
#define DD   256     // in channels
#define FF   256     // filter channels
#define TMAX 4096    // mel_max_length
#define EPSS 1e-5f

// workspace layout (bytes)
#define OFF_XH   (0)                       // x as f16:  16*512*256*2 = 4 MiB
#define OFF_H1H  (4194304)                 // h1 as f16: 4 MiB
#define OFF_B1   (8388608)                 // conv1 W repacked: 393216 B
#define OFF_B2   (8781824)                 // conv2 W repacked: 393216 B
#define OFF_CUM  (9175040)                 // cumsum: 16*512*4 = 32 KiB

// ---------------- f32 -> f16 convert (vectorized) ----------------
__global__ void f32_to_f16_kernel(const float* __restrict__ in,
                                  _Float16* __restrict__ out, int n4) {
  int i = blockIdx.x * blockDim.x + threadIdx.x;
  if (i < n4) {
    float4 v = ((const float4*)in)[i];
    union { _Float16 h[4]; unsigned long long u; } p;
    p.h[0] = (_Float16)v.x; p.h[1] = (_Float16)v.y;
    p.h[2] = (_Float16)v.z; p.h[3] = (_Float16)v.w;
    ((unsigned long long*)out)[i] = p.u;
  }
}

// ---- repack conv weight [F][D][3] into per-lane WMMA B fragments ----
// Bpack[((t*8+c)*16 + ftile)*512 + lane*16 + h] = w[f=ftile*16+(lane&15)]
//                                                  [d=c*32 + (lane>>4)*16 + h][t]
// (B pack uses contiguous k, not AMD's table order: lanes 0-15 hold K=0..15, lanes 16-31 hold K=16..31)
__global__ void repack_w_kernel(const float* __restrict__ w,
                                _Float16* __restrict__ bp) {
  int e = blockIdx.x * 256 + threadIdx.x;          // 0 .. 196607
  int h     =  e        & 15;
  int lane  = (e >> 4)  & 31;
  int ftile = (e >> 9)  & 15;
  int c     = (e >> 13) & 7;
  int t     =  e >> 16;                            // 0..2
  int s  = lane >> 4, nn = lane & 15;
  int f  = ftile * 16 + nn;
  int d  = c * 32 + s * 16 + h;
  bp[e] = (_Float16)w[(f * DD + d) * 3 + t];
}

// ---------------- per-batch cumsum of durations (1 wave32 each) ----------------
__global__ void cumsum_kernel(const int* __restrict__ tgt, int* __restrict__ cum) {
  int n = blockIdx.x, lane = threadIdx.x;          // 32 lanes, 16 elems each
  int base = n * LL + lane * 16;
  int v[16]; int run = 0;
  #pragma unroll
  for (int i = 0; i < 16; ++i) { run += tgt[base + i]; v[i] = run; }
  int incl = run;
  #pragma unroll
  for (int off = 1; off < 32; off <<= 1) {
    int u = __shfl_up(incl, off);
    if (lane >= off) incl += u;
  }
  int excl = incl - run;
  #pragma unroll
  for (int i = 0; i < 16; ++i) cum[base + i] = excl + v[i];
}

// ---------------- fused conv1d(K=3) + bias + LayerNorm + ReLU ----------------
// FINAL=false: write f16 activations.  FINAL=true: fuse F->1 linear -> dpo.
template <bool FINAL>
__global__ __launch_bounds__(512)
void conv_ln_kernel(const _Float16* __restrict__ inp,     // [N,L,256] f16
                    const _Float16* __restrict__ bpack,   // repacked weights
                    const float* __restrict__ cbias,
                    const float* __restrict__ lng, const float* __restrict__ lnb,
                    const float* __restrict__ lw,  const float* __restrict__ lbp,
                    _Float16* __restrict__ outh, float* __restrict__ dpo) {
  __shared__ _Float16 shX[18 * 256];    // rows l0-1 .. l0+16
  __shared__ float    shOut[16 * 256];

  const int n  = blockIdx.y;
  const int l0 = blockIdx.x * 16;
  const int tid = threadIdx.x;

  // stage 18 input rows (zero-padded at sequence edges), 16B chunks
  for (int ch = tid; ch < 18 * 32; ch += 512) {
    int r = ch >> 5;
    int col = (ch & 31) * 8;
    int l = l0 + r - 1;
    v4u val = {0u, 0u, 0u, 0u};
    if (l >= 0 && l < LL)
      val = *(const v4u*)(inp + (size_t)(n * LL + l) * 256 + col);
    *(v4u*)(&shX[r * 256 + col]) = val;
  }
  __syncthreads();

  const int wv   = tid >> 5;      // wave 0..15 -> channel tile f = wv*16..
  const int lane = tid & 31;
  const int s = lane >> 4;        // lane half-group
  const int m = lane & 15;        // A-matrix row within tile

  v8f acc = {};
  #pragma unroll
  for (int t = 0; t < 3; ++t) {
    #pragma unroll
    for (int c = 0; c < 8; ++c) {
      union { v16h h; v4u u[2]; } A, B;
      const int arow = m + t;           // l = l0 + m + t - 1
      const int base = c * 32;
      // A 16x32 layout (matches B pack): halfs 0-15 -> K = base+s*16..base+s*16+15
      A.u[0] = *(const v4u*)(&shX[arow * 256 + base + s * 16]);
      A.u[1] = *(const v4u*)(&shX[arow * 256 + base + s * 16 + 8]);
      const _Float16* bp = bpack + (size_t)(((t * 8 + c) * 16 + wv) * 32 + lane) * 16;
      B.u[0] = ((const v4u*)bp)[0];
      B.u[1] = ((const v4u*)bp)[1];
      acc = __builtin_amdgcn_wmma_f32_16x16x32_f16(false, A.h, false, B.h,
                                                   (short)0, acc, false, false);
    }
  }

  // bias + scatter tile to LDS (C/D layout: M = r + 8*s, N = lane&15)
  const float bias = cbias[wv * 16 + m];
  #pragma unroll
  for (int r = 0; r < 8; ++r) {
    int M = r + 8 * s;
    shOut[M * 256 + wv * 16 + m] = acc[r] + bias;
  }
  __syncthreads();

  // LayerNorm over 256 channels: wave wv owns row wv (wave32 shuffle reduce)
  float s1 = 0.f, s2 = 0.f, vals[8];
  #pragma unroll
  for (int t = 0; t < 8; ++t) {
    float v = shOut[wv * 256 + lane + 32 * t];
    vals[t] = v; s1 += v; s2 += v * v;
  }
  #pragma unroll
  for (int off = 16; off > 0; off >>= 1) {
    s1 += __shfl_xor(s1, off);
    s2 += __shfl_xor(s2, off);
  }
  const float mean = s1 * (1.f / 256.f);
  const float var  = s2 * (1.f / 256.f) - mean * mean;
  const float rs   = rsqrtf(var + EPSS);

  if (!FINAL) {
    #pragma unroll
    for (int t = 0; t < 8; ++t) {
      int f = lane + 32 * t;
      float hn = fmaxf((vals[t] - mean) * rs * lng[f] + lnb[f], 0.f);
      outh[(size_t)(n * LL + l0 + wv) * 256 + f] = (_Float16)hn;
    }
  } else {
    float dot = 0.f;
    #pragma unroll
    for (int t = 0; t < 8; ++t) {
      int f = lane + 32 * t;
      float hn = fmaxf((vals[t] - mean) * rs * lng[f] + lnb[f], 0.f);
      dot += hn * lw[f];
    }
    #pragma unroll
    for (int off = 16; off > 0; off >>= 1) dot += __shfl_xor(dot, off);
    if (lane == 0) dpo[n * LL + l0 + wv] = fmaxf(dot + lbp[0], 0.f);
  }
}

// ---------------- length regulator: searchsorted gather ----------------
__global__ __launch_bounds__(256)
void length_reg_kernel(const float* __restrict__ x, const int* __restrict__ cum,
                       float* __restrict__ out) {
  __shared__ int c[LL];
  const int n = blockIdx.y;
  for (int i = threadIdx.x; i < LL; i += 256) c[i] = cum[n * LL + i];
  __syncthreads();

  const int t = blockIdx.x * 16 + (threadIdx.x >> 4);
  const int q = threadIdx.x & 15;
  int lo = 0, hi = LL;
  while (lo < hi) {                     // searchsorted(cum, t, side='right')
    int mid = (lo + hi) >> 1;
    if (c[mid] <= t) lo = mid + 1; else hi = mid;
  }
  const bool valid = t < c[LL - 1];
  const int src = lo < LL ? lo : LL - 1;
  const float4* xs = (const float4*)(x + ((size_t)n * LL + src) * DD);
  float4* od = (float4*)(out + ((size_t)n * TMAX + t) * DD);
  const float4 z = make_float4(0.f, 0.f, 0.f, 0.f);
  #pragma unroll
  for (int j = 0; j < 4; ++j) {
    int qq = q + 16 * j;
    od[qq] = valid ? xs[qq] : z;
  }
}

extern "C" void kernel_launch(void* const* d_in, const int* in_sizes, int n_in,
                              void* d_out, int out_size, void* d_ws, size_t ws_size,
                              hipStream_t stream) {
  const float* x   = (const float*)d_in[0];
  const float* c1w = (const float*)d_in[1];
  const float* c1b = (const float*)d_in[2];
  const float* g1  = (const float*)d_in[3];
  const float* b1  = (const float*)d_in[4];
  const float* c2w = (const float*)d_in[5];
  const float* c2b = (const float*)d_in[6];
  const float* g2  = (const float*)d_in[7];
  const float* b2  = (const float*)d_in[8];
  const float* lw  = (const float*)d_in[9];
  const float* lb  = (const float*)d_in[10];
  const int*   tgt = (const int*)d_in[11];

  char* ws = (char*)d_ws;
  _Float16* xh  = (_Float16*)(ws + OFF_XH);
  _Float16* h1h = (_Float16*)(ws + OFF_H1H);
  _Float16* bp1 = (_Float16*)(ws + OFF_B1);
  _Float16* bp2 = (_Float16*)(ws + OFF_B2);
  int*      cum = (int*)(ws + OFF_CUM);

  float* out = (float*)d_out;                         // [16,4096,256]
  float* dpo = out + (size_t)NB * TMAX * DD;          // [16,512]

  f32_to_f16_kernel<<<4096, 256, 0, stream>>>(x, xh, NB * LL * DD / 4);
  repack_w_kernel<<<768, 256, 0, stream>>>(c1w, bp1);
  repack_w_kernel<<<768, 256, 0, stream>>>(c2w, bp2);
  cumsum_kernel<<<NB, 32, 0, stream>>>(tgt, cum);

  conv_ln_kernel<false><<<dim3(LL / 16, NB), 512, 0, stream>>>(
      xh, bp1, c1b, g1, b1, nullptr, nullptr, h1h, nullptr);
  conv_ln_kernel<true><<<dim3(LL / 16, NB), 512, 0, stream>>>(
      h1h, bp2, c2b, g2, b2, lw, lb, nullptr, dpo);

  length_reg_kernel<<<dim3(TMAX / 16, NB), 256, 0, stream>>>(x, cum, out);
}
